// LogicGatedSNN_81381040325547
// MI455X (gfx1250) — matrix operand order repaired and checked by reading, not compiled, not run on hardware
//
#include <hip/hip_runtime.h>

typedef __attribute__((ext_vector_type(16))) _Float16 v16h;
typedef __attribute__((ext_vector_type(8)))  float    v8f;
typedef __attribute__((ext_vector_type(4)))  float    v4f;   // native vector for NT builtins

#define IN      8192
#define OUT     8192
#define THRESH  50.0f
#define KSPLIT  8
#define KSLICE  (IN / KSPLIT)   // 1024

// -------------------------------------------------------------------------
// Kernel 1: ternarize(states) @ spike via v_wmma_f32_16x16x32_f16, split-K.
// Each wave handles a 16-row group over a 1024-wide K slice; partial sums
// are written to part[ks*OUT + row] (no atomics -> bitwise deterministic).
// -------------------------------------------------------------------------
__global__ __launch_bounds__(256) void snn_gemv_wmma(
    const float* __restrict__ states,
    const float* __restrict__ spike,
    float* __restrict__ part)
{
    __shared__ __align__(32) _Float16 ssp[KSLICE];   // spike slice as f16 0/1

    const int tid  = threadIdx.x;
    const int ks   = blockIdx.y;
    const int kofs = ks * KSLICE;

    for (int i = tid; i < KSLICE; i += 256)
        ssp[i] = (spike[kofs + i] > 0.5f) ? (_Float16)1.0f : (_Float16)0.0f;
    __syncthreads();

    const int wave = tid >> 5;
    const int lane = tid & 31;
    const int m    = lane & 15;      // row within 16-row group (both halves)
    const int half = lane >> 4;      // selects contiguous 16-wide K sub-block (k = 16*half + s)

    const int    row_base = blockIdx.x * 128 + wave * 16;
    const float* rowp     = states + (size_t)(row_base + m) * IN + kofs;

    v8f c = {};
    for (int kk = 0; kk < KSLICE; kk += 32) {
        const int ka = kk + half * 16;       // K = ka .. ka+7
        const int kb = ka + 8;               // K = kb .. kb+7
        v4f f0 = __builtin_nontemporal_load((const v4f*)(rowp + ka));
        v4f f1 = __builtin_nontemporal_load((const v4f*)(rowp + ka + 4));
        v4f f2 = __builtin_nontemporal_load((const v4f*)(rowp + kb));
        v4f f3 = __builtin_nontemporal_load((const v4f*)(rowp + kb + 4));

        const float fa[16] = { f0.x, f0.y, f0.z, f0.w,  f1.x, f1.y, f1.z, f1.w,
                               f2.x, f2.y, f2.z, f2.w,  f3.x, f3.y, f3.z, f3.w };
        v16h a;
#pragma unroll
        for (int i = 0; i < 16; ++i)
            a[i] = (fa[i] > THRESH) ? (_Float16)1.0f : (_Float16)0.0f;

        // B 32x16: every N column = spike value for that K (replicated),
        // so each lane's 16 K-values are ssp[kk + half*16 + 0..15].
        v16h b = *(const v16h*)(ssp + kk + half * 16);

        c = __builtin_amdgcn_wmma_f32_16x16x32_f16(
                /*neg_a=*/false, a, /*neg_b=*/false, b,
                /*c_mod=*/(short)0, c, /*reuse_a=*/false, /*reuse_b=*/false);
    }

    // C/D layout: VGPR r, lanes 0-15 -> M=r (N=lane), lanes 16-31 -> M=8+r.
    // All N columns identical; lanes with N==0 (lane 0 and 16) write out.
    if ((lane & 15) == 0) {
        const int mb = half * 8;
#pragma unroll
        for (int r = 0; r < 8; ++r)
            part[(size_t)ks * OUT + row_base + mb + r] = c[r];
    }
}

// -------------------------------------------------------------------------
// Kernel 2: reduce split-K partials (fixed order) + neuron state update.
// -------------------------------------------------------------------------
__global__ __launch_bounds__(256) void snn_neuron(
    const float* __restrict__ part,
    const float* __restrict__ v_mem,
    const float* __restrict__ v_th,
    const float* __restrict__ noise,
    float* __restrict__ out_spikes,
    float* __restrict__ out_vmem,
    float* __restrict__ out_vth)
{
    const int o = blockIdx.x * 256 + threadIdx.x;
    if (o >= OUT) return;

    float current = 0.0f;
#pragma unroll
    for (int k = 0; k < KSPLIT; ++k)
        current += part[(size_t)k * OUT + o];

    const float vth   = v_th[o];
    const float v_new = v_mem[o] * 0.8f + current + noise[o];
    const float s     = (v_new >= vth) ? 1.0f : 0.0f;

    out_spikes[o] = s;
    out_vth[o]    = fminf(fmaxf(vth + (s - 0.05f) * 0.1f, 0.5f), 10.0f);
    out_vmem[o]   = v_new * (1.0f - s) * 0.2f;
}

// -------------------------------------------------------------------------
// Kernel 3: eligibility trace, 512 MB streaming pass.
// One row per blockIdx.y so the postsynaptic spike is a scalar; v4f +
// non-temporal hints keep the one-shot elig traffic out of L2.
// -------------------------------------------------------------------------
__global__ __launch_bounds__(256) void snn_elig(
    const float* __restrict__ elig,
    const float* __restrict__ spike,
    const float* __restrict__ spikes_o,
    float* __restrict__ out_elig)
{
    const int o  = blockIdx.y;
    const int i4 = blockIdx.x * 256 + threadIdx.x;         // float4 index
    const float  so   = spikes_o[o];
    const size_t base = (size_t)o * (IN / 4) + i4;

    const v4f e  = __builtin_nontemporal_load(((const v4f*)elig) + base);
    const v4f sp = *(((const v4f*)spike) + i4);            // hot, cacheable

    v4f r;
    r.x = fminf(fmaxf(e.x * 0.95f + so * sp.x, 0.0f), 5.0f);
    r.y = fminf(fmaxf(e.y * 0.95f + so * sp.y, 0.0f), 5.0f);
    r.z = fminf(fmaxf(e.z * 0.95f + so * sp.z, 0.0f), 5.0f);
    r.w = fminf(fmaxf(e.w * 0.95f + so * sp.w, 0.0f), 5.0f);

    __builtin_nontemporal_store(r, ((v4f*)out_elig) + base);
}

// -------------------------------------------------------------------------
// Host launcher. Inputs (reference order): spike_input, states, v_mem, v_th,
// elig, noise. Outputs concatenated: spikes[O], v_mem_new[O], v_th_new[O],
// elig_new[O*I].
// -------------------------------------------------------------------------
extern "C" void kernel_launch(void* const* d_in, const int* in_sizes, int n_in,
                              void* d_out, int out_size, void* d_ws, size_t ws_size,
                              hipStream_t stream) {
    const float* spike  = (const float*)d_in[0];
    const float* states = (const float*)d_in[1];
    const float* v_mem  = (const float*)d_in[2];
    const float* v_th   = (const float*)d_in[3];
    const float* elig   = (const float*)d_in[4];
    const float* noise  = (const float*)d_in[5];

    float* out        = (float*)d_out;
    float* out_spikes = out;
    float* out_vmem   = out + OUT;
    float* out_vth    = out + 2 * OUT;
    float* out_elig   = out + 3 * OUT;

    float* part = (float*)d_ws;                    // KSPLIT * OUT floats (256 KB)

    dim3 g1(OUT / 128, KSPLIT);                    // 64 x 8 blocks, 8 waves each
    snn_gemv_wmma<<<g1, 256, 0, stream>>>(states, spike, part);

    snn_neuron<<<OUT / 256, 256, 0, stream>>>(part, v_mem, v_th, noise,
                                              out_spikes, out_vmem, out_vth);

    dim3 g3(IN / (4 * 256), OUT);                  // 8 x 8192 blocks
    snn_elig<<<g3, 256, 0, stream>>>(elig, spike, out_spikes, out_elig);
}
